// ViT_UNet_5085241278492
// MI455X (gfx1250) — matrix operand order repaired and checked by reading, not compiled, not run on hardware
//
#include <hip/hip_runtime.h>
#include <hip/hip_bf16.h>
#include <math.h>

// ---------------- CDNA5 WMMA types ----------------
typedef __attribute__((ext_vector_type(16))) __bf16       v16bf;
typedef __attribute__((ext_vector_type(8)))  float        v8f;
typedef __attribute__((ext_vector_type(8)))  unsigned int v8u;

#define B_   8
#define N_   256
#define C_   3072
#define H_   8
#define HD_  384
#define FF_  12288
#define EPS_ 1e-5f
#define LPITCH 40   // LDS row pitch in ushorts (80B: 16B-aligned rows for async b128)

__device__ __forceinline__ unsigned short bf16_bits(float a) {
    return __builtin_bit_cast(unsigned short, (__bf16)a);
}

// ---- CDNA5 async global->LDS copy (ASYNCcnt-tracked), 16 bytes per lane ----
__device__ __forceinline__ void async_ld_b128(unsigned int lds_byte_addr, const void* gaddr) {
    asm volatile("global_load_async_to_lds_b128 %0, %1, off"
                 :: "v"(lds_byte_addr), "v"((unsigned long long)gaddr)
                 : "memory");
}
__device__ __forceinline__ void wait_async0() {
    asm volatile("s_wait_asynccnt 0" ::: "memory");
}

// ================= per-patch 3x3 conv QKV (fp32 in, bf16 out) =================
// q,k head-split (B,H,N,hd); v written TRANSPOSED (B,H,hd,N) so attn@V's B-operand
// is row-major per output column (same LDS path as Q@K^T).
__global__ __launch_bounds__(256) void conv_qkv(
    const float* __restrict__ x,
    const float* __restrict__ wq, const float* __restrict__ wk, const float* __restrict__ wv,
    __bf16* __restrict__ q, __bf16* __restrict__ k, __bf16* __restrict__ vT)
{
    __shared__ float lX[3072];
    __shared__ float lW[3][81];
    const int bn  = blockIdx.x;          // b*N + n
    const int tid = threadIdx.x;
    const float* xp = x + (long)bn * 3072;
    for (int t = tid; t < 3072; t += 256) lX[t] = xp[t];
    if (tid < 81) { lW[0][tid] = wq[tid]; lW[1][tid] = wk[tid]; lW[2][tid] = wv[tid]; }
    __syncthreads();
    const int b = bn >> 8, n = bn & 255;
    for (int j = 0; j < 12; ++j) {
        const int o   = j * 256 + tid;        // flat channel index (co,py,px)
        const int co  = o >> 10;
        const int rem = o & 1023;
        const int py  = rem >> 5, px = rem & 31;
        float aq = 0.f, ak = 0.f, av = 0.f;
        #pragma unroll
        for (int ci = 0; ci < 3; ++ci)
            #pragma unroll
            for (int ky = 0; ky < 3; ++ky) {
                const int yy = py + ky - 1;
                if ((unsigned)yy >= 32u) continue;
                #pragma unroll
                for (int kx = 0; kx < 3; ++kx) {
                    const int xx = px + kx - 1;
                    if ((unsigned)xx >= 32u) continue;
                    const float xin = lX[ci * 1024 + yy * 32 + xx];
                    const int wi = ((co * 3 + ci) * 3 + ky) * 3 + kx;
                    aq += xin * lW[0][wi];
                    ak += xin * lW[1][wi];
                    av += xin * lW[2][wi];
                }
            }
        const int h = o / HD_, d = o - h * HD_;
        const long bh = (long)b * H_ + h;
        const long oi = (bh * N_ + n) * HD_ + d;       // (B,H,N,hd)
        q[oi] = (__bf16)aq; k[oi] = (__bf16)ak;
        vT[(bh * HD_ + d) * N_ + n] = (__bf16)av;       // (B,H,hd,N)
    }
}

// ================= generic tiled WMMA GEMM, 128x128x32, 8 wave32s =================
// LDS layout (both operands): [row][k] with pitch LPITCH.
//  - A rows = output rows (m); global A is (M x K) bf16 row-major.
//  - B rows = output cols (n); BBF16: global Bt is (Nn x K) bf16 row-major (async path).
//                              !BBF16: global B is (K x Nn) fp32; transposed while converting.
enum { EPI_SCORE = 0, EPI_ATTNV = 1, EPI_PROJ = 2, EPI_FFN1 = 3, EPI_FFN2 = 4 };

template<int EPI, bool BBF16, int ZDIV>
__global__ __launch_bounds__(256) void gemm_wmma(
    const __bf16* __restrict__ A, long aBatch, int lda,
    const void*   __restrict__ Bg, long bBatch, int ldb,
    const float*  __restrict__ bias,
    const float*  __restrict__ resid,
    float* __restrict__ outF, __bf16* __restrict__ outB,
    long obO, long obI, int ldo,
    int M, int Nn, int K)
{
    __shared__ unsigned short sA[128 * LPITCH];
    __shared__ unsigned short sB[128 * LPITCH];

    const int tid  = threadIdx.x;
    const int lane = tid & 31;
    const int wv   = tid >> 5;
    const int lm   = lane & 15;
    const int lh   = lane >> 4;
    const int mW   = (wv >> 2) * 64;   // 2 wave-rows
    const int nW   = (wv & 3) * 32;    // 4 wave-cols

    const int  z     = blockIdx.z;
    const long mBase = (long)blockIdx.y * 128;
    const long nBase = (long)blockIdx.x * 128;

    const __bf16* Az = A + (long)z * aBatch;
    const unsigned int sAb = (unsigned int)(unsigned long long)&sA[0];
    const unsigned int sBb = (unsigned int)(unsigned long long)&sB[0];

    v8f acc[4][2] = {};

    for (int kt = 0; kt < K; kt += 32) {
        // ---- stage A tile (128 x 32 bf16): async DMA, 2 threads/row, 2x b128 each ----
        {
            const int r = tid >> 1, hf = tid & 1;
            const __bf16* src = Az + (mBase + r) * (long)lda + kt + hf * 16;
            const unsigned int l = sAb + (unsigned int)(r * LPITCH + hf * 16) * 2u;
            async_ld_b128(l,      src);
            async_ld_b128(l + 16, (const char*)src + 16);
        }
        // ---- stage B tile ----
        if constexpr (BBF16) {   // Bt (Nn x K) bf16 row-major -> async to sB[n][k]
            const __bf16* Bt = (const __bf16*)Bg + (long)z * bBatch;
            const int r = tid >> 1, hf = tid & 1;
            const __bf16* src = Bt + (nBase + r) * (long)ldb + kt + hf * 16;
            const unsigned int l = sBb + (unsigned int)(r * LPITCH + hf * 16) * 2u;
            async_ld_b128(l,      src);
            async_ld_b128(l + 16, (const char*)src + 16);
        } else {                 // B (K x Nn) fp32 -> convert + transpose -> sB[n][k]
            const float* Bp = (const float*)Bg + (long)z * bBatch;
            const int r = tid >> 3, cs = tid & 7;          // r: k-row 0..31, cs: col-seg
            const float* srcf = Bp + (kt + r) * (long)ldb + nBase + cs * 16;
            const float4* s4 = (const float4*)srcf;
            if (kt + 32 < K)  // prefetch next K-tile of the weights
                __builtin_prefetch((const void*)(srcf + (long)ldb * 32), 0, 1);
            #pragma unroll
            for (int j = 0; j < 4; ++j) {
                const float4 f = s4[j];
                const int nc = cs * 16 + 4 * j;
                sB[(nc + 0) * LPITCH + r] = bf16_bits(f.x);
                sB[(nc + 1) * LPITCH + r] = bf16_bits(f.y);
                sB[(nc + 2) * LPITCH + r] = bf16_bits(f.z);
                sB[(nc + 3) * LPITCH + r] = bf16_bits(f.w);
            }
        }
        wait_async0();
        __syncthreads();

        // ---- build fragments (pair loads; LDS rows 4B-aligned at even k) ----
        v8u afr[4];
        #pragma unroll
        for (int mt = 0; mt < 4; ++mt) {
            const unsigned short* rp = &sA[(mW + mt * 16 + lm) * LPITCH];
            #pragma unroll
            for (int i = 0; i < 8; ++i) {
                const int k0 = lh * 8 + (i < 4 ? 2 * i : 16 + 2 * (i - 4));
                afr[mt][i] = *(const unsigned int*)&rp[k0];
            }
        }
        v8u bfr[2];
        #pragma unroll
        for (int nt = 0; nt < 2; ++nt) {
            const unsigned short* rp = &sB[(nW + nt * 16 + lm) * LPITCH];
            #pragma unroll
            for (int i = 0; i < 8; ++i) {
                const int k0 = lh * 8 + (i < 4 ? 2 * i : 16 + 2 * (i - 4));
                bfr[nt][i] = *(const unsigned int*)&rp[k0];
            }
        }
        // ---- 8 WMMAs per wave per K-step ----
        #pragma unroll
        for (int mt = 0; mt < 4; ++mt)
            #pragma unroll
            for (int nt = 0; nt < 2; ++nt)
                acc[mt][nt] = __builtin_amdgcn_wmma_f32_16x16x32_bf16(
                    false, __builtin_bit_cast(v16bf, afr[mt]),
                    false, __builtin_bit_cast(v16bf, bfr[nt]),
                    (short)0, acc[mt][nt], false, false);
        __syncthreads();
    }

    // ---- epilogue (ZDIV is compile-time: shifts, no integer-div sequence) ----
    const long obase = (long)(z / ZDIV) * obO + (long)(z % ZDIV) * obI;
    #pragma unroll
    for (int mt = 0; mt < 4; ++mt) {
        #pragma unroll
        for (int nt = 0; nt < 2; ++nt) {
            #pragma unroll
            for (int i = 0; i < 8; ++i) {
                const long row = mBase + mW + mt * 16 + lh * 8 + i;
                const long col = nBase + nW + nt * 16 + lm;
                const long idx = obase + row * (long)ldo + col;
                const float val = acc[mt][nt][i];
                if constexpr (EPI == EPI_SCORE) {
                    outF[idx] = val;
                } else if constexpr (EPI == EPI_ATTNV) {
                    outB[idx] = (__bf16)val;
                } else if constexpr (EPI == EPI_PROJ) {
                    outF[idx] = val + bias[col] + resid[idx];
                } else if constexpr (EPI == EPI_FFN1) {
                    const float t = val + bias[col];
                    outB[idx] = (__bf16)(0.5f * t * (1.0f + erff(t * 0.70710678118654752f)));
                } else { // EPI_FFN2
                    outF[idx] = val + bias[col] + resid[idx];
                }
            }
        }
    }
}

// ================= softmax over rows of 256 (scale folded in) =================
__global__ __launch_bounds__(256) void softmax_rows(float* __restrict__ S, float scale)
{
    __shared__ float red[256];
    float* p = S + (long)blockIdx.x * 256;
    const int t = threadIdx.x;
    const float vv = p[t] * scale;
    red[t] = vv; __syncthreads();
    for (int s = 128; s > 0; s >>= 1) { if (t < s) red[t] = fmaxf(red[t], red[t + s]); __syncthreads(); }
    const float mx = red[0]; __syncthreads();
    const float e = __expf(vv - mx);
    red[t] = e; __syncthreads();
    for (int s = 128; s > 0; s >>= 1) { if (t < s) red[t] += red[t + s]; __syncthreads(); }
    p[t] = e * (1.0f / red[0]);
}

// ================= re-attention head mix + BatchNorm(eval) -> bf16 =================
__global__ __launch_bounds__(256) void mix_bn(
    const float* __restrict__ S, const float* __restrict__ w, const float* __restrict__ bw,
    const float* __restrict__ g, const float* __restrict__ bb,
    const float* __restrict__ mean, const float* __restrict__ var,
    __bf16* __restrict__ out)
{
    const long idx = (long)blockIdx.x * 256 + threadIdx.x;  // over B*N*N
    const long b = idx >> 16;
    const long r = idx & 65535;
    float si[H_];
    #pragma unroll
    for (int i = 0; i < H_; ++i) si[i] = S[(b * H_ + i) * 65536 + r];
    #pragma unroll
    for (int o = 0; o < H_; ++o) {
        float a = bw[o];
        #pragma unroll
        for (int i = 0; i < H_; ++i) a += w[o * H_ + i] * si[i];
        a = (a - mean[o]) * (g[o] * rsqrtf(var[o] + EPS_)) + bb[o];
        out[(b * H_ + o) * 65536 + r] = (__bf16)a;
    }
}

// ================= LayerNorm over (N,C) per batch =================
__global__ __launch_bounds__(256) void ln_part(const float* __restrict__ z, float* __restrict__ partials)
{
    const int b = blockIdx.y, chunk = blockIdx.x;  // 64 chunks of 12288
    const float* p = z + (long)b * (N_ * C_) + (long)chunk * 12288;
    float s = 0.f, s2 = 0.f;
    for (int t = threadIdx.x; t < 12288; t += 256) { const float v = p[t]; s += v; s2 += v * v; }
    __shared__ float rs[256], rq[256];
    rs[threadIdx.x] = s; rq[threadIdx.x] = s2; __syncthreads();
    for (int k = 128; k > 0; k >>= 1) {
        if (threadIdx.x < k) { rs[threadIdx.x] += rs[threadIdx.x + k]; rq[threadIdx.x] += rq[threadIdx.x + k]; }
        __syncthreads();
    }
    if (threadIdx.x == 0) {
        partials[(b * 64 + chunk) * 2]     = rs[0];
        partials[(b * 64 + chunk) * 2 + 1] = rq[0];
    }
}

__global__ void ln_stats(const float* __restrict__ partials, float* __restrict__ stats)
{
    const int b = blockIdx.x;
    __shared__ float rs[64], rq[64];
    const int t = threadIdx.x;  // blockDim = 64
    rs[t] = partials[(b * 64 + t) * 2];
    rq[t] = partials[(b * 64 + t) * 2 + 1];
    __syncthreads();
    for (int k = 32; k > 0; k >>= 1) {
        if (t < k) { rs[t] += rs[t + k]; rq[t] += rq[t + k]; }
        __syncthreads();
    }
    if (t == 0) {
        const float inv_n = 1.0f / (float)(N_ * C_);
        const float mu  = rs[0] * inv_n;
        const float var = rq[0] * inv_n - mu * mu;
        stats[b * 2] = mu; stats[b * 2 + 1] = rsqrtf(var + EPS_);
    }
}

__global__ __launch_bounds__(256) void ln_apply(
    const float* __restrict__ z, const float* __restrict__ stats,
    const float* __restrict__ g, const float* __restrict__ bt,
    float* __restrict__ outF, __bf16* __restrict__ outB)
{
    const int  b   = blockIdx.y;
    const long r   = (long)blockIdx.x * 256 + threadIdx.x;  // 0 .. N*C-1
    const long idx = (long)b * (N_ * C_) + r;
    const float y = (z[idx] - stats[b * 2]) * stats[b * 2 + 1] * g[r] + bt[r];
    if (outF) outF[idx] = y;
    if (outB) outB[idx] = (__bf16)y;
}

// ================= launch =================
extern "C" void kernel_launch(void* const* d_in, const int* in_sizes, int n_in,
                              void* d_out, int out_size, void* d_ws, size_t ws_size,
                              hipStream_t stream)
{
    (void)in_sizes; (void)n_in; (void)out_size; (void)ws_size;
    const float* x        = (const float*)d_in[0];
    const float* qconv_w  = (const float*)d_in[1];
    const float* kconv_w  = (const float*)d_in[2];
    const float* vconv_w  = (const float*)d_in[3];
    const float* reat_w   = (const float*)d_in[4];
    const float* reat_b   = (const float*)d_in[5];
    const float* bn_g     = (const float*)d_in[6];
    const float* bn_b     = (const float*)d_in[7];
    const float* bn_m     = (const float*)d_in[8];
    const float* bn_v     = (const float*)d_in[9];
    const float* proj_w   = (const float*)d_in[10];
    const float* proj_b   = (const float*)d_in[11];
    const float* ln_g     = (const float*)d_in[12];
    const float* ln_b     = (const float*)d_in[13];
    const float* ff_w1    = (const float*)d_in[14];
    const float* ff_b1    = (const float*)d_in[15];
    const float* ff_w2    = (const float*)d_in[16];
    const float* ff_b2    = (const float*)d_in[17];

    const long BNC  = (long)B_ * N_ * C_;      // 6,291,456
    const long BHNN = (long)B_ * H_ * N_ * N_; // 4,194,304

    char* ws = (char*)d_ws;
    auto alloc = [&](size_t bytes) -> char* {
        char* p = ws; ws += (bytes + 255) & ~(size_t)255; return p;
    };
    __bf16* q       = (__bf16*)alloc(BNC * 2);
    __bf16* kbuf    = (__bf16*)alloc(BNC * 2);
    __bf16* vT      = (__bf16*)alloc(BNC * 2);
    float*  S       = (float*)alloc(BHNN * 4);
    __bf16* attnb   = (__bf16*)alloc(BHNN * 2);
    __bf16* attnout = (__bf16*)alloc(BNC * 2);
    float*  z1      = (float*)alloc(BNC * 4);
    float*  y1      = (float*)alloc(BNC * 4);
    __bf16* y1b     = (__bf16*)alloc(BNC * 2);
    float*  partials = (float*)alloc(B_ * 64 * 2 * 4);
    float*  stats    = (float*)alloc(B_ * 2 * 4);
    // aliases over dead buffers (exact fits, byte-verified):
    float*  z2 = (float*)q;     // q+k region  = 2*BNC*2 = BNC*4 bytes
    __bf16* h  = (__bf16*)vT;   // vT+S+attnb+attnout region = B*N*FF*2 bytes

    // 1. conv QKV -> bf16; q,k head-split (B,H,N,hd); v transposed (B,H,hd,N)
    conv_qkv<<<B_ * N_, 256, 0, stream>>>(x, qconv_w, kconv_w, vconv_w, q, kbuf, vT);

    // 2. S = Q @ K^T  (batched over B*H; bf16 async path)
    gemm_wmma<EPI_SCORE, true, 1><<<dim3(2, 2, B_ * H_), 256, 0, stream>>>(
        q, (long)N_ * HD_, HD_, kbuf, (long)N_ * HD_, HD_,
        nullptr, nullptr, S, nullptr,
        (long)N_ * N_, 0, N_, N_, N_, HD_);

    // 3. softmax with scale = hd^-0.5
    softmax_rows<<<B_ * H_ * N_, 256, 0, stream>>>(S, 0.05103103630798288f);

    // 4. head mix (8x8) + BN -> bf16 attention
    mix_bn<<<(unsigned)(B_ * N_ * N_ / 256), 256, 0, stream>>>(
        S, reat_w, reat_b, bn_g, bn_b, bn_m, bn_v, attnb);

    // 5. out = attn @ V (V pre-transposed), scatter heads into (B,N,C) bf16
    gemm_wmma<EPI_ATTNV, true, H_><<<dim3(3, 2, B_ * H_), 256, 0, stream>>>(
        attnb, (long)N_ * N_, N_, vT, (long)HD_ * N_, N_,
        nullptr, nullptr, nullptr, attnout,
        (long)N_ * C_, HD_, C_, N_, HD_, N_);

    // 6. z1 = x + attnout @ proj_w + proj_b
    gemm_wmma<EPI_PROJ, false, 1><<<dim3(24, 16, 1), 256, 0, stream>>>(
        attnout, 0, C_, proj_w, 0, C_,
        proj_b, x, z1, nullptr,
        0, 0, C_, B_ * N_, C_, C_);

    // 7. LN1 -> y1 (fp32) + y1b (bf16)
    ln_part<<<dim3(64, B_), 256, 0, stream>>>(z1, partials);
    ln_stats<<<B_, 64, 0, stream>>>(partials, stats);
    ln_apply<<<dim3(N_ * C_ / 256, B_), 256, 0, stream>>>(z1, stats, ln_g, ln_b, y1, y1b);

    // 8. h = gelu(y1 @ ff_w1 + b1)  (bf16)
    gemm_wmma<EPI_FFN1, false, 1><<<dim3(96, 16, 1), 256, 0, stream>>>(
        y1b, 0, C_, ff_w1, 0, FF_,
        ff_b1, nullptr, nullptr, h,
        0, 0, FF_, B_ * N_, FF_, C_);

    // 9. z2 = y1 + h @ ff_w2 + b2
    gemm_wmma<EPI_FFN2, false, 1><<<dim3(24, 16, 1), 256, 0, stream>>>(
        h, 0, FF_, ff_w2, 0, C_,
        ff_b2, y1, z2, nullptr,
        0, 0, C_, B_ * N_, C_, FF_);

    // 10. LN2 -> d_out (fp32)
    ln_part<<<dim3(64, B_), 256, 0, stream>>>(z2, partials);
    ln_stats<<<B_, 64, 0, stream>>>(partials, stats);
    ln_apply<<<dim3(N_ * C_ / 256, B_), 256, 0, stream>>>(z2, stats, ln_g, ln_b, (float*)d_out, nullptr);
}
